// stochastic_RNN_autoreg_torchscript_38293928411397
// MI455X (gfx1250) — matrix operand order repaired and checked by reading, not compiled, not run on hardware
//
#include <hip/hip_runtime.h>
#include <hip/hip_bf16.h>
#include <math.h>

// ---------------------------------------------------------------------------
// Model dims
// ---------------------------------------------------------------------------
#define NLEV  60
#define NX    4
#define NH    64
#define NMEM  64
#define NSFC  17
#define NY    4
#define BATCH 4096

typedef __attribute__((ext_vector_type(16))) __bf16 v16bf;
typedef __attribute__((ext_vector_type(8)))  float  v8f;

__device__ __forceinline__ v8f wmma_bf16(v16bf a, v16bf b, v8f c) {
  // (neg_a, A, neg_b, B, c_mod, C, reuse_a, reuse_b)
  return __builtin_amdgcn_wmma_f32_16x16x32_bf16(false, a, false, b, (short)0, c,
                                                 false, false);
}

// Branch-free activations: v_exp_f32 + v_rcp_f32 only (no EXEC divergence).
__device__ __forceinline__ float fast_rcp(float x) {
  return __builtin_amdgcn_rcpf(x);
}
__device__ __forceinline__ float sigmoidf_(float x) {
  return fast_rcp(1.0f + __expf(-x));
}
__device__ __forceinline__ float tanh_fast(float x) {
  // 1 - 2/(exp(2x)+1); saturates to +/-1 via IEEE inf/0 semantics.
  return 1.0f - 2.0f * fast_rcp(1.0f + __expf(2.0f * x));
}

// A-matrix fragment (16x32 bf16) from a row-major bf16 buffer (LDS or global).
// Lanes 0-15 hold K = k0+{0..7, 16..23}; lanes 16-31 hold K = k0+{8..15, 24..31}.
__device__ __forceinline__ v16bf load_a_bf16(const __bf16* base, int row_stride,
                                             int k0, int lane) {
  const int m    = lane & 15;
  const int koff = k0 + ((lane & 16) ? 8 : 0);
  const __bf16* p = base + (size_t)m * row_stride + koff;
  v16bf a;
#pragma unroll
  for (int e = 0; e < 8; ++e) { a[e] = p[e]; a[8 + e] = p[16 + e]; }
  return a;
}

// B-matrix fragment (32x16 bf16) gathered from row-major fp32 W[ldw cols].
// Lanes 0-15: K = k0+{0..7, 16..23} ; lanes 16-31: K = k0+{8..15, 24..31} (col N = lane&15).
__device__ __forceinline__ v16bf gather_b(const float* W, int ldw, int k0, int n0,
                                          int lane) {
  const int n  = n0 + (lane & 15);
  const int kb = k0 + ((lane & 16) ? 8 : 0);
  v16bf b;
#pragma unroll
  for (int e = 0; e < 16; ++e) b[e] = (__bf16)W[(size_t)(kb + e + (e & 8)) * ldw + n];
  return b;
}

// ---------------------------------------------------------------------------
// Kernel P: preprocessing (fully parallel, bandwidth-bound).
//  - rnn1_mem (B,60,64) fp32  -> membf (60,B,64) bf16  (transpose + convert)
//  - inputs_main (B,60,4) f32 -> xlev (60,B,4) f32     (transpose)
// ---------------------------------------------------------------------------
__global__ void __launch_bounds__(256)
prep_kernel(const float* __restrict__ rnn1_mem,
            const float* __restrict__ inputs_main,
            __bf16* __restrict__ membf,
            float* __restrict__ xlev) {
  const int row = blockIdx.x * blockDim.x + threadIdx.x;  // lev*B + b
  if (row >= NLEV * BATCH) return;
  const int lev = row / BATCH;
  const int b   = row % BATCH;
  const float* src = rnn1_mem + ((size_t)b * NLEV + lev) * NMEM;
  __bf16* dst = membf + (size_t)row * NMEM;
#pragma unroll
  for (int j = 0; j < NMEM; j += 4) {
    const float4 v = *(const float4*)(src + j);
    dst[j + 0] = (__bf16)v.x;
    dst[j + 1] = (__bf16)v.y;
    dst[j + 2] = (__bf16)v.z;
    dst[j + 3] = (__bf16)v.w;
  }
  *(float4*)(xlev + (size_t)row * NX) =
      *(const float4*)(inputs_main + ((size_t)b * NLEV + lev) * NX);
}

// ---------------------------------------------------------------------------
// Kernel 0: initial hidden/cell states (tiny GEMMs, VALU; accurate tanhf)
// ---------------------------------------------------------------------------
__global__ void __launch_bounds__(256)
init_states_kernel(const float* __restrict__ aux,     // (B,17)
                   const float* __restrict__ W_sfc,  const float* __restrict__ b_sfc,
                   const float* __restrict__ W_sfc2, const float* __restrict__ b_sfc2,
                   const float* __restrict__ W_toa,  const float* __restrict__ b_toa,
                   const float* __restrict__ W_toa2, const float* __restrict__ b_toa2,
                   float* __restrict__ h0_1, float* __restrict__ c0_1,
                   float* __restrict__ h0_2, float* __restrict__ c0_2) {
  const int tid = blockIdx.x * blockDim.x + threadIdx.x;
  if (tid >= BATCH * NH) return;
  const int b = tid >> 6, j = tid & 63;
  const float* a = aux + (size_t)b * NSFC;
  float sh = b_sfc[j], sc = b_sfc2[j];
#pragma unroll
  for (int k = 0; k < NSFC; ++k) {
    const float av = a[k];
    sh += av * W_sfc[k * NH + j];
    sc += av * W_sfc2[k * NH + j];
  }
  h0_1[tid] = tanhf(sh);
  c0_1[tid] = tanhf(sc);
  const float t0 = a[1], t1 = a[6];
  h0_2[tid] = tanhf(t0 * W_toa[j]  + t1 * W_toa[NH + j]  + b_toa[j]);
  c0_2[tid] = tanhf(t0 * W_toa2[j] + t1 * W_toa2[NH + j] + b_toa2[j]);
}

// ---------------------------------------------------------------------------
// Kernel 1: rnn1 scan (bottom-up).  Block = 16 batch rows, 4 waves.
// Wave w owns gate columns [16w,16w+16) of i/f/g/o and of logvar.
// gates = x@Wx1[0:4](K=4, VALU) + mem@Wx1[4:](K=64 WMMA) + h@Wh1(K=64 WMMA)
// h is double-buffered in LDS (one barrier per step).
// ---------------------------------------------------------------------------
__global__ void __launch_bounds__(128)
rnn1_scan_kernel(const __bf16* __restrict__ membf, // (60,B,64) bf16
                 const float* __restrict__ xlev,   // (60,B,4)  f32
                 const float* __restrict__ eps1,   // (60,B,64)
                 const float* __restrict__ h0, const float* __restrict__ c0,
                 const float* __restrict__ Wx1,   // (68,256)
                 const float* __restrict__ Wh1,   // (64,256)
                 const float* __restrict__ Wxs1,  // (68,64)
                 const float* __restrict__ Whs1,  // (64,64)
                 __bf16* __restrict__ rnn1out) {  // (60,B,64) bf16
  const int tid  = threadIdx.x;
  const int lane = tid & 31;
  const int wave = tid >> 5;
  const int b0   = blockIdx.x * 16;
  const int hi   = (lane >> 4) & 1;
  const int ncol = lane & 15;
  const int jcol = wave * 16 + ncol;  // hidden column owned by this lane

  __shared__ __align__(16) __bf16 hbuf[2][16 * 72];

  // seed h into hbuf[0]
  for (int idx = tid; idx < 16 * NH; idx += 128) {
    const int r = idx >> 6, j = idx & 63;
    hbuf[0][r * 72 + j] = (__bf16)h0[(size_t)(b0 + r) * NH + j];
  }

  // c slice in C/D layout: element r -> row r+8*hi, col jcol
  v8f c;
#pragma unroll
  for (int r = 0; r < 8; ++r)
    c[r] = c0[(size_t)(b0 + r + 8 * hi) * NH + jcol];

  // preload B fragments (weights), bf16 in registers
  v16bf Bh[4][2], Bxm[4][2], Bhs[2], Bxs[2];
#pragma unroll
  for (int t = 0; t < 4; ++t) {
    const int n0 = t * 64 + wave * 16;
#pragma unroll
    for (int kc = 0; kc < 2; ++kc) {
      Bh[t][kc]  = gather_b(Wh1, 4 * NH, kc * 32, n0, lane);
      Bxm[t][kc] = gather_b(Wx1 + (size_t)NX * 4 * NH, 4 * NH, kc * 32, n0, lane);
    }
  }
#pragma unroll
  for (int kc = 0; kc < 2; ++kc) {
    Bhs[kc] = gather_b(Whs1, NH, kc * 32, wave * 16, lane);
    Bxs[kc] = gather_b(Wxs1 + (size_t)NX * NH, NH, kc * 32, wave * 16, lane);
  }

  // x-part (K=4) weights for VALU accumulation
  float wx[4][4], wxs[4];
#pragma unroll
  for (int t = 0; t < 4; ++t)
#pragma unroll
    for (int kk = 0; kk < 4; ++kk)
      wx[t][kk] = Wx1[kk * 4 * NH + t * 64 + jcol];
#pragma unroll
  for (int kk = 0; kk < 4; ++kk) wxs[kk] = Wxs1[kk * NH + jcol];

  // streaming pointers (advance by fixed strides each step)
  const float* epsl = eps1 + (size_t)(b0 + 8 * hi) * NH + jcol;           // +BATCH*NH
  __bf16* outl = rnn1out +
      ((size_t)(NLEV - 1) * BATCH + b0 + 8 * hi) * NH + jcol;             // -BATCH*NH

  __syncthreads();

  int cur = 0;
  for (int step = 0; step < NLEV; ++step) {
    const int lev = NLEV - 1 - step;  // bottom-up

    // ---- mem-side: A fragments (bf16, level-major, contiguous tile) ----
    const __bf16* memb = membf + ((size_t)lev * BATCH + b0) * NMEM;
    const v16bf Am0 = load_a_bf16(memb, NMEM, 0, lane);
    const v16bf Am1 = load_a_bf16(memb, NMEM, 32, lane);

    // ---- K=4 raw-x part via VALU, one row live at a time ----
    const float* xrow = xlev + ((size_t)lev * BATCH + b0 + 8 * hi) * NX;
    v8f acc[4], lv;
#pragma unroll
    for (int r = 0; r < 8; ++r) {
      const float4 xv = *(const float4*)(xrow + (size_t)r * NX);
#pragma unroll
      for (int t = 0; t < 4; ++t)
        acc[t][r] = xv.x * wx[t][0] + xv.y * wx[t][1] +
                    xv.z * wx[t][2] + xv.w * wx[t][3];
      lv[r] = xv.x * wxs[0] + xv.y * wxs[1] + xv.z * wxs[2] + xv.w * wxs[3];
    }

    // mem-side WMMAs
#pragma unroll
    for (int t = 0; t < 4; ++t) {
      acc[t] = wmma_bf16(Am0, Bxm[t][0], acc[t]);
      acc[t] = wmma_bf16(Am1, Bxm[t][1], acc[t]);
    }
    lv = wmma_bf16(Am0, Bxs[0], lv);
    lv = wmma_bf16(Am1, Bxs[1], lv);

    // ---- h-side: A fragments from current LDS buffer ----
    const v16bf Ah0 = load_a_bf16(hbuf[cur], 72, 0, lane);
    const v16bf Ah1 = load_a_bf16(hbuf[cur], 72, 32, lane);
#pragma unroll
    for (int t = 0; t < 4; ++t) {
      acc[t] = wmma_bf16(Ah0, Bh[t][0], acc[t]);
      acc[t] = wmma_bf16(Ah1, Bh[t][1], acc[t]);
    }
    lv = wmma_bf16(Ah0, Bhs[0], lv);
    lv = wmma_bf16(Ah1, Bhs[1], lv);

    // ---- lane-local LSTM update + reparameterized noise (branch-free) ----
    const int nxt = cur ^ 1;
#pragma unroll
    for (int r = 0; r < 8; ++r) {
      const float ep = epsl[(size_t)r * NH];
      const float iv = sigmoidf_(acc[0][r]);
      const float fv = sigmoidf_(acc[1][r]);
      const float gv = tanh_fast(acc[2][r]);
      const float ov = sigmoidf_(acc[3][r]);
      const float cn = fv * c[r] + iv * gv;
      const float hn = ov * tanh_fast(cn) + ep * __expf(0.5f * lv[r]);
      c[r] = cn;
      const __bf16 hb = (__bf16)hn;
      hbuf[nxt][(r + 8 * hi) * 72 + jcol] = hb;
      outl[(size_t)r * NH] = hb;  // stored at level `lev` (flip back)
    }
    epsl += (size_t)BATCH * NH;
    outl -= (size_t)BATCH * NH;
    cur = nxt;
    __syncthreads();  // new h visible for next step
  }
}

// ---------------------------------------------------------------------------
// Kernel 2: rnn2 scan (top-down).  x = rnn1out (bf16, level-major).
// ---------------------------------------------------------------------------
__global__ void __launch_bounds__(128)
rnn2_scan_kernel(const __bf16* __restrict__ rnn1out, // (60,B,64) bf16
                 const float* __restrict__ eps2,     // (60,B,64)
                 const float* __restrict__ h0, const float* __restrict__ c0,
                 const float* __restrict__ Wx2,   // (64,256)
                 const float* __restrict__ Wh2,   // (64,256)
                 const float* __restrict__ Wxs2,  // (64,64)
                 const float* __restrict__ Whs2,  // (64,64)
                 __bf16* __restrict__ rnn2out,    // (60,B,64) bf16
                 float* __restrict__ hlast) {     // (B,64) fp32
  const int tid  = threadIdx.x;
  const int lane = tid & 31;
  const int wave = tid >> 5;
  const int b0   = blockIdx.x * 16;
  const int hi   = (lane >> 4) & 1;
  const int ncol = lane & 15;
  const int jcol = wave * 16 + ncol;

  __shared__ __align__(16) __bf16 hbuf[2][16 * 72];

  for (int idx = tid; idx < 16 * NH; idx += 128) {
    const int r = idx >> 6, j = idx & 63;
    hbuf[0][r * 72 + j] = (__bf16)h0[(size_t)(b0 + r) * NH + j];
  }

  v8f c;
#pragma unroll
  for (int r = 0; r < 8; ++r)
    c[r] = c0[(size_t)(b0 + r + 8 * hi) * NH + jcol];

  v16bf Bh[4][2], Bx[4][2], Bhs[2], Bxs[2];
#pragma unroll
  for (int t = 0; t < 4; ++t) {
    const int n0 = t * 64 + wave * 16;
#pragma unroll
    for (int kc = 0; kc < 2; ++kc) {
      Bh[t][kc] = gather_b(Wh2, 4 * NH, kc * 32, n0, lane);
      Bx[t][kc] = gather_b(Wx2, 4 * NH, kc * 32, n0, lane);
    }
  }
#pragma unroll
  for (int kc = 0; kc < 2; ++kc) {
    Bhs[kc] = gather_b(Whs2, NH, kc * 32, wave * 16, lane);
    Bxs[kc] = gather_b(Wxs2, NH, kc * 32, wave * 16, lane);
  }

  const float* epsl = eps2 + (size_t)(b0 + 8 * hi) * NH + jcol;   // +BATCH*NH
  __bf16* outl = rnn2out + (size_t)(b0 + 8 * hi) * NH + jcol;     // +BATCH*NH
  const __bf16* xbase = rnn1out + (size_t)b0 * NH;                // +BATCH*NH

  __syncthreads();

  int cur = 0;
  for (int step = 0; step < NLEV; ++step) {
    const v16bf Ax0 = load_a_bf16(xbase, NH, 0, lane);
    const v16bf Ax1 = load_a_bf16(xbase, NH, 32, lane);

    v8f acc[4] = {v8f{}, v8f{}, v8f{}, v8f{}};
    v8f lv = {};
#pragma unroll
    for (int t = 0; t < 4; ++t) {
      acc[t] = wmma_bf16(Ax0, Bx[t][0], acc[t]);
      acc[t] = wmma_bf16(Ax1, Bx[t][1], acc[t]);
    }
    lv = wmma_bf16(Ax0, Bxs[0], lv);
    lv = wmma_bf16(Ax1, Bxs[1], lv);

    const v16bf Ah0 = load_a_bf16(hbuf[cur], 72, 0, lane);
    const v16bf Ah1 = load_a_bf16(hbuf[cur], 72, 32, lane);
#pragma unroll
    for (int t = 0; t < 4; ++t) {
      acc[t] = wmma_bf16(Ah0, Bh[t][0], acc[t]);
      acc[t] = wmma_bf16(Ah1, Bh[t][1], acc[t]);
    }
    lv = wmma_bf16(Ah0, Bhs[0], lv);
    lv = wmma_bf16(Ah1, Bhs[1], lv);

    const int nxt = cur ^ 1;
#pragma unroll
    for (int r = 0; r < 8; ++r) {
      const float ep = epsl[(size_t)r * NH];
      const float iv = sigmoidf_(acc[0][r]);
      const float fv = sigmoidf_(acc[1][r]);
      const float gv = tanh_fast(acc[2][r]);
      const float ov = sigmoidf_(acc[3][r]);
      const float cn = fv * c[r] + iv * gv;
      const float hn = ov * tanh_fast(cn) + ep * __expf(0.5f * lv[r]);
      c[r] = cn;
      const __bf16 hb = (__bf16)hn;
      hbuf[nxt][(r + 8 * hi) * 72 + jcol] = hb;
      outl[(size_t)r * NH] = hb;
      if (step == NLEV - 1)
        hlast[(size_t)(b0 + 8 * hi + r) * NH + jcol] = hn;
    }
    epsl  += (size_t)BATCH * NH;
    outl  += (size_t)BATCH * NH;
    xbase += (size_t)BATCH * NH;
    cur = nxt;
    __syncthreads();
  }
}

// ---------------------------------------------------------------------------
// Kernel 3: latent head.  latent = rnn2out @ W_lat + b_lat (WMMA),
// new_mem = latent, out = latent @ W_out + b_out (VALU via LDS).
// ---------------------------------------------------------------------------
__global__ void __launch_bounds__(128)
head_kernel(const __bf16* __restrict__ rnn2out, // (60,B,64)
            const float* __restrict__ W_lat, const float* __restrict__ b_lat,
            const float* __restrict__ W_out, const float* __restrict__ b_out,
            float* __restrict__ out,        // (B,60,4)
            float* __restrict__ new_mem) {  // (B,60,64)
  const int tid  = threadIdx.x;
  const int lane = tid & 31;
  const int wave = tid >> 5;
  const int R0   = blockIdx.x * 16;  // row index into (l*B + b)
  const int l    = R0 / BATCH;
  const int b0   = R0 % BATCH;
  const int hi   = (lane >> 4) & 1;
  const int ncol = lane & 15;

  __shared__ __align__(16) float lat[16][68];

  const __bf16* ab = rnn2out + (size_t)R0 * NH;
  const v16bf A0 = load_a_bf16(ab, NH, 0, lane);
  const v16bf A1 = load_a_bf16(ab, NH, 32, lane);
  const v16bf B0 = gather_b(W_lat, NMEM, 0, wave * 16, lane);
  const v16bf B1 = gather_b(W_lat, NMEM, 32, wave * 16, lane);
  v8f acc = {};
  acc = wmma_bf16(A0, B0, acc);
  acc = wmma_bf16(A1, B1, acc);

  const int j = wave * 16 + ncol;
  const float bj = b_lat[j];
#pragma unroll
  for (int r = 0; r < 8; ++r) {
    const int mrow = r + 8 * hi;
    const float v  = acc[r] + bj;
    lat[mrow][j] = v;
    new_mem[((size_t)(b0 + mrow) * NLEV + l) * NMEM + j] = v;
  }
  __syncthreads();

  if (tid < 64) {
    const int mrow = tid >> 2, y = tid & 3;
    float s = b_out[y];
#pragma unroll
    for (int jj = 0; jj < NMEM; ++jj) s += lat[mrow][jj] * W_out[jj * NY + y];
    out[((size_t)(b0 + mrow) * NLEV + l) * NY + y] = s;
  }
}

// ---------------------------------------------------------------------------
// Kernel 4: surface heads on last hidden (tiny, VALU)
// ---------------------------------------------------------------------------
__global__ void __launch_bounds__(256)
sfc_head_kernel(const float* __restrict__ hlast,   // (B,64)
                const float* __restrict__ eps_sfc, // (B,2)
                const float* __restrict__ W_sfcout, const float* __restrict__ b_sfcout,
                const float* __restrict__ W_mu, const float* __restrict__ b_mu,
                const float* __restrict__ W_lv, const float* __restrict__ b_lv,
                float* __restrict__ out_sfc) {     // (B,3)
  const int b = blockIdx.x * blockDim.x + threadIdx.x;
  if (b >= BATCH) return;
  const float* h = hlast + (size_t)b * NH;
  float rad = b_sfcout[0];
  float mu0 = b_mu[0], mu1 = b_mu[1];
  float l0 = b_lv[0], l1 = b_lv[1];
#pragma unroll
  for (int j = 0; j < NH; ++j) {
    const float hv = h[j];
    rad += hv * W_sfcout[j];
    mu0 += hv * W_mu[j * 2 + 0];
    mu1 += hv * W_mu[j * 2 + 1];
    l0  += hv * W_lv[j * 2 + 0];
    l1  += hv * W_lv[j * 2 + 1];
  }
  out_sfc[(size_t)b * 3 + 0] = mu0 + eps_sfc[b * 2 + 0] * __expf(0.5f * l0);
  out_sfc[(size_t)b * 3 + 1] = mu1 + eps_sfc[b * 2 + 1] * __expf(0.5f * l1);
  out_sfc[(size_t)b * 3 + 2] = rad;
}

// ---------------------------------------------------------------------------
// Host launcher
// ---------------------------------------------------------------------------
extern "C" void kernel_launch(void* const* d_in, const int* in_sizes, int n_in,
                              void* d_out, int out_size, void* d_ws, size_t ws_size,
                              hipStream_t stream) {
  const float* inputs_main = (const float*)d_in[0];
  const float* inputs_aux  = (const float*)d_in[1];
  const float* rnn1_mem    = (const float*)d_in[2];
  const float* eps1        = (const float*)d_in[3];
  const float* eps2        = (const float*)d_in[4];
  const float* eps_sfc     = (const float*)d_in[5];
  const float* W_sfc   = (const float*)d_in[6];
  const float* b_sfc   = (const float*)d_in[7];
  const float* W_sfc2  = (const float*)d_in[8];
  const float* b_sfc2  = (const float*)d_in[9];
  const float* W_toa   = (const float*)d_in[10];
  const float* b_toa   = (const float*)d_in[11];
  const float* W_toa2  = (const float*)d_in[12];
  const float* b_toa2  = (const float*)d_in[13];
  const float* Wx1     = (const float*)d_in[14];
  const float* Wh1     = (const float*)d_in[15];
  const float* Wxs1    = (const float*)d_in[16];
  const float* Whs1    = (const float*)d_in[17];
  const float* Wx2     = (const float*)d_in[18];
  const float* Wh2     = (const float*)d_in[19];
  const float* Wxs2    = (const float*)d_in[20];
  const float* Whs2    = (const float*)d_in[21];
  const float* W_lat   = (const float*)d_in[22];
  const float* b_lat   = (const float*)d_in[23];
  const float* W_out   = (const float*)d_in[24];
  const float* b_out   = (const float*)d_in[25];
  const float* W_sfcout = (const float*)d_in[26];
  const float* b_sfcout = (const float*)d_in[27];
  const float* W_mu    = (const float*)d_in[28];
  const float* b_mu    = (const float*)d_in[29];
  const float* W_lv    = (const float*)d_in[30];
  const float* b_lv    = (const float*)d_in[31];

  // d_out: out (B,60,4) | out_sfc (B,3) | new_mem (B,60,64)
  float* out_p    = (float*)d_out;
  float* outsfc_p = out_p + (size_t)BATCH * NLEV * NY;
  float* newmem_p = outsfc_p + (size_t)BATCH * 3;

  // workspace partition
  char* ws = (char*)d_ws;
  size_t off = 0;
  auto take = [&](size_t bytes) -> void* {
    void* p = ws + off;
    off = (off + bytes + 255) & ~(size_t)255;
    return p;
  };
  float* h0_1 = (float*)take((size_t)BATCH * NH * sizeof(float));
  float* c0_1 = (float*)take((size_t)BATCH * NH * sizeof(float));
  float* h0_2 = (float*)take((size_t)BATCH * NH * sizeof(float));
  float* c0_2 = (float*)take((size_t)BATCH * NH * sizeof(float));
  __bf16* membf   = (__bf16*)take((size_t)NLEV * BATCH * NMEM * sizeof(__bf16));
  float*  xlev    = (float*)take((size_t)NLEV * BATCH * NX * sizeof(float));
  __bf16* rnn1out = (__bf16*)take((size_t)NLEV * BATCH * NH * sizeof(__bf16));
  __bf16* rnn2out = (__bf16*)take((size_t)NLEV * BATCH * NH * sizeof(__bf16));
  float* hlast = (float*)take((size_t)BATCH * NH * sizeof(float));
  (void)ws_size; (void)in_sizes; (void)n_in; (void)out_size;

  prep_kernel<<<(NLEV * BATCH + 255) / 256, 256, 0, stream>>>(
      rnn1_mem, inputs_main, membf, xlev);

  init_states_kernel<<<(BATCH * NH) / 256, 256, 0, stream>>>(
      inputs_aux, W_sfc, b_sfc, W_sfc2, b_sfc2, W_toa, b_toa, W_toa2, b_toa2,
      h0_1, c0_1, h0_2, c0_2);

  rnn1_scan_kernel<<<BATCH / 16, 128, 0, stream>>>(
      membf, xlev, eps1, h0_1, c0_1, Wx1, Wh1, Wxs1, Whs1, rnn1out);

  rnn2_scan_kernel<<<BATCH / 16, 128, 0, stream>>>(
      rnn1out, eps2, h0_2, c0_2, Wx2, Wh2, Wxs2, Whs2, rnn2out, hlast);

  head_kernel<<<(NLEV * BATCH) / 16, 128, 0, stream>>>(
      rnn2out, W_lat, b_lat, W_out, b_out, out_p, newmem_p);

  sfc_head_kernel<<<BATCH / 256, 256, 0, stream>>>(
      hlast, eps_sfc, W_sfcout, b_sfcout, W_mu, b_mu, W_lv, b_lv, outsfc_p);
}
